// Semantic2FeatureBlock_67284957659469
// MI455X (gfx1250) — compile-verified, matrix operand order checked by reading, not run on hardware
//
#include <hip/hip_runtime.h>
#include <math.h>

// ---------------------------------------------------------------- types
typedef __bf16 bf16_t;
typedef __attribute__((ext_vector_type(16))) __bf16 v16bf;
typedef __attribute__((ext_vector_type(8)))  __bf16 v8bf;
typedef __attribute__((ext_vector_type(8)))  float  v8f;
typedef __attribute__((ext_vector_type(4)))  float  f32x4;
typedef __attribute__((ext_vector_type(4)))  unsigned int u32x4;
typedef __attribute__((ext_vector_type(8)))  unsigned int u32x8;

#define C_DIM   768
#define CLS_DIM 20
#define B_DIM   16
#define N_PIX   1024
#define HEADS   4
#define DH      192
#define C2      1536
#define EPSF    1e-5f

// ---------------------------------------------------------------- wmma helper
__device__ inline v8f wmma_bf16(v16bf a, v16bf b, v8f c) {
    return __builtin_amdgcn_wmma_f32_16x16x32_bf16(
        /*neg_a=*/false, a, /*neg_b=*/false, b,
        /*c_mod=*/(short)0, c, /*reuse_a=*/false, /*reuse_b=*/false);
}

__device__ inline v16bf mk16(v8bf lo, v8bf hi) {
    v16bf r;
#pragma unroll
    for (int i = 0; i < 8; ++i) { r[i] = lo[i]; r[i + 8] = hi[i]; }
    return r;
}

// A/B fragment, K-contiguous in memory (row-major along K), bf16 source.
// 16-bit A-matrix 16x32 layout: lane m=l%16; elems 0..7 -> K = kh..kh+7,
// elems 8..15 -> K = 16+kh..; kh = (l/16)*8.
__device__ inline v16bf fragK_bf16(const bf16_t* __restrict__ base, long ld,
                                   int row0, int k0, int lane) {
    const bf16_t* p = base + (long)(row0 + (lane & 15)) * ld + k0 + ((lane >> 4) << 3);
    return mk16(*(const v8bf*)p, *(const v8bf*)(p + 16));
}
__device__ inline v16bf fragK_bf16_clamp(const bf16_t* __restrict__ base, long ld,
                                         int row0, int rowmax, int k0, int lane) {
    int row = row0 + (lane & 15); if (row > rowmax) row = rowmax;
    const bf16_t* p = base + (long)row * ld + k0 + ((lane >> 4) << 3);
    return mk16(*(const v8bf*)p, *(const v8bf*)(p + 16));
}
// K-contiguous fragment, fp32 source converted on the fly.
__device__ inline v16bf fragK_f32(const float* __restrict__ base, long ld,
                                  int row0, int k0, int lane) {
    const float* p = base + (long)(row0 + (lane & 15)) * ld + k0 + ((lane >> 4) << 3);
    v16bf r;
#pragma unroll
    for (int i = 0; i < 8; ++i) { r[i] = (bf16_t)p[i]; r[i + 8] = (bf16_t)p[16 + i]; }
    return r;
}
// K-strided fragment (column-major along K), bf16 / f32 sources.
__device__ inline v16bf fragS_bf16(const bf16_t* __restrict__ base, long ld,
                                   int col0, int k0, int lane) {
    const bf16_t* p = base + (long)(k0 + ((lane >> 4) << 3)) * ld + col0 + (lane & 15);
    v16bf r;
#pragma unroll
    for (int i = 0; i < 8; ++i) { r[i] = p[(long)i * ld]; r[i + 8] = p[(long)(i + 16) * ld]; }
    return r;
}
__device__ inline v16bf fragS_f32(const float* __restrict__ base, long ld,
                                  int col0, int k0, int lane) {
    const float* p = base + (long)(k0 + ((lane >> 4) << 3)) * ld + col0 + (lane & 15);
    v16bf r;
#pragma unroll
    for (int i = 0; i < 8; ++i) { r[i] = (bf16_t)p[(long)i * ld]; r[i + 8] = (bf16_t)p[(long)(i + 16) * ld]; }
    return r;
}

// ---------------------------------------------------------------- 1. batch stats
__global__ __launch_bounds__(256) void stats_partial_kernel(
    const float* __restrict__ x, float* __restrict__ part) {
    int b = blockIdx.y, seg = blockIdx.x;               // 32 segments x 24576 elems
    const float* p = x + (size_t)b * (C_DIM * N_PIX) + (size_t)seg * 24576;
    float s = 0.f, ss = 0.f;
    for (int i = threadIdx.x; i < 24576; i += 256) { float v = p[i]; s += v; ss += v * v; }
    __shared__ float s1[256], s2[256];
    s1[threadIdx.x] = s; s2[threadIdx.x] = ss; __syncthreads();
    for (int d = 128; d > 0; d >>= 1) {
        if (threadIdx.x < d) { s1[threadIdx.x] += s1[threadIdx.x + d]; s2[threadIdx.x] += s2[threadIdx.x + d]; }
        __syncthreads();
    }
    if (threadIdx.x == 0) {
        part[(b * 32 + seg) * 2 + 0] = s1[0];
        part[(b * 32 + seg) * 2 + 1] = s2[0];
    }
}
__global__ __launch_bounds__(32) void stats_final_kernel(
    const float* __restrict__ part, float* __restrict__ stats) {
    int b = blockIdx.x;
    if (threadIdx.x == 0) {
        float s = 0.f, ss = 0.f;
        for (int i = 0; i < 32; ++i) { s += part[(b * 32 + i) * 2]; ss += part[(b * 32 + i) * 2 + 1]; }
        const float inv_n = 1.0f / (float)(C_DIM * N_PIX);
        float mu = s * inv_n;
        float var = ss * inv_n - mu * mu;
        stats[2 * b] = mu;
        stats[2 * b + 1] = rsqrtf(var + EPSF);
    }
}

// ---------------------------------------------------------------- 2. LN(x_cls) -> bf16
__global__ __launch_bounds__(256) void ln_cls_kernel(
    const float* __restrict__ x, const float* __restrict__ w,
    const float* __restrict__ bb, bf16_t* __restrict__ o) {
    int row = blockIdx.x;                                // 0..319
    const float* p = x + (size_t)row * C_DIM;
    int tid = threadIdx.x;
    float a0 = p[tid], a1 = p[tid + 256], a2 = p[tid + 512];
    __shared__ float sh[256];
    sh[tid] = a0 + a1 + a2; __syncthreads();
    for (int d = 128; d > 0; d >>= 1) { if (tid < d) sh[tid] += sh[tid + d]; __syncthreads(); }
    float mu = sh[0] * (1.0f / C_DIM);
    __syncthreads();
    float c0 = a0 - mu, c1 = a1 - mu, c2 = a2 - mu;
    sh[tid] = c0 * c0 + c1 * c1 + c2 * c2; __syncthreads();
    for (int d = 128; d > 0; d >>= 1) { if (tid < d) sh[tid] += sh[tid + d]; __syncthreads(); }
    float rstd = rsqrtf(sh[0] * (1.0f / C_DIM) + EPSF);
    bf16_t* op = o + (size_t)row * C_DIM;
    op[tid]       = (bf16_t)(c0 * rstd * w[tid]       + bb[tid]);
    op[tid + 256] = (bf16_t)(c1 * rstd * w[tid + 256] + bb[tid + 256]);
    op[tid + 512] = (bf16_t)(c2 * rstd * w[tid + 512] + bb[tid + 512]);
}

// ---------------------------------------------------------------- 3. q = LNx @ Wq^T + bq  (scale folded in)
__global__ __launch_bounds__(32) void q_gemm_kernel(
    const bf16_t* __restrict__ lnx, const float* __restrict__ Wq,
    const float* __restrict__ bq, bf16_t* __restrict__ qout) {
    int jt = blockIdx.x * 16, mt = blockIdx.y * 16;
    int lane = threadIdx.x;
    v8f acc = {};
    for (int k0 = 0; k0 < C_DIM; k0 += 32) {
        v16bf a = fragK_bf16(lnx, C_DIM, mt, k0, lane);
        v16bf b = fragK_f32(Wq, C_DIM, jt, k0, lane);    // B[k,j] = Wq[j,k]: K-contiguous
        acc = wmma_bf16(a, b, acc);
    }
    const float scale = 0.07216878364870322f;            // 1/sqrt(192)
    int col = jt + (lane & 15);
    float bias = bq[col];
#pragma unroll
    for (int r = 0; r < 8; ++r) {
        int row = mt + r + ((lane >> 4) << 3);
        qout[(size_t)row * C_DIM + col] = (bf16_t)((acc[r] + bias) * scale);
    }
}

// ---------------------------------------------------------------- 4. kv = Wkv @ nx
// Dominant GEMM. A-tile (Wkv, pure data movement) is brought in by the
// Tensor Data Mover (tensor_load_to_lds + TENSORcnt); B-tile needs the fused
// LayerNorm affine + bf16 convert, so it goes through the VALU path into LDS.
__global__ __launch_bounds__(256) void kv_gemm_kernel(
    const float* __restrict__ xpatch, const float* __restrict__ Wkv,
    const float* __restrict__ bkv, const float* __restrict__ nxw,
    const float* __restrict__ nxb, const float* __restrict__ stats,
    bf16_t* __restrict__ kv) {
    __shared__ __align__(16) float  Asf[64][36];         // fp32 A tile, TDM pads 4 DW / 32 DW
    __shared__ __align__(16) bf16_t Bs[64][40];          // B^T[n][k] bf16 (K-contiguous)
    int b  = blockIdx.z;
    int n0 = blockIdx.x * 64;
    int m0 = blockIdx.y * 64;
    float mu = stats[2 * b], rs = stats[2 * b + 1];
    int tid = threadIdx.x, lane = tid & 31, wave = tid >> 5;
    int wm = wave >> 1, wn = wave & 1;                   // 4x2 wave grid, each 16x32
    const float* xb = xpatch + (size_t)b * C_DIM * N_PIX;
    unsigned lds_a = (unsigned)(unsigned long long)(const void*)&Asf[0][0];
    v8f acc0 = {}, acc1 = {};
    for (int k0 = 0; k0 < C_DIM; k0 += 32) {
        if (tid < 32) {  // wave 0 drives the TDM for the 64x32-dword A tile
            unsigned long long ga =
                (unsigned long long)(const void*)(Wkv + (size_t)m0 * C_DIM + k0);
            u32x4 g0;
            g0[0] = 1u;                                              // count=1, user D#
            g0[1] = __builtin_amdgcn_readfirstlane(lds_a);           // lds_addr
            g0[2] = __builtin_amdgcn_readfirstlane((unsigned)ga);    // global_addr[31:0]
            g0[3] = __builtin_amdgcn_readfirstlane(
                        (unsigned)((ga >> 32) & 0x01ffffffu)) | (2u << 30); // [56:32] | type=2
            u32x8 g1;
            g1[0] = (2u << 16) | (1u << 20) | (4u << 22) | (3u << 25); // 4B elems, pad 4DW/32DW
            g1[1] = (unsigned)(C_DIM & 0xffffu) << 16;               // tensor_dim0 lo16
            g1[2] = (unsigned)(C_DIM >> 16) | ((unsigned)(C2 & 0xffffu) << 16); // dim0 hi | dim1 lo
            g1[3] = (unsigned)(C2 >> 16) | (32u << 16);              // dim1 hi | tile_dim0=32
            g1[4] = 64u;                                             // tile_dim1=64, tile_dim2=0
            g1[5] = (unsigned)C_DIM;                                 // tensor_dim0_stride lo32
            g1[6] = 0u;
            g1[7] = 0u;
            asm volatile("tensor_load_to_lds %0, %1" :: "s"(g0), "s"(g1) : "memory");
            __builtin_amdgcn_s_wait_tensorcnt(0);
        }
        {   // B tile: 32 k x 64 n, thread -> 8 contiguous fp32 along n, LN fused
            int kk = tid >> 3, nn = (tid & 7) * 8;
            int c = k0 + kk;
            const float* src = xb + (size_t)c * N_PIX + n0 + nn;
            if (k0 + 32 < C_DIM) __builtin_prefetch(src + 32 * N_PIX, 0, 1);
            float sc = rs * nxw[c];
            float bi = nxb[c] - mu * sc;
#pragma unroll
            for (int i = 0; i < 8; ++i) Bs[nn + i][kk] = (bf16_t)(src[i] * sc + bi);
        }
        __syncthreads();
        {
            int arow = wm * 16 + (lane & 15);
            int kh = (lane >> 4) << 3;
            const float* ar = &Asf[arow][kh];
            f32x4 a0 = *(const f32x4*)ar;
            f32x4 a1 = *(const f32x4*)(ar + 4);
            f32x4 a2 = *(const f32x4*)(ar + 16);
            f32x4 a3 = *(const f32x4*)(ar + 20);
            v16bf af;
#pragma unroll
            for (int i = 0; i < 4; ++i) {
                af[i]      = (bf16_t)a0[i];
                af[i + 4]  = (bf16_t)a1[i];
                af[i + 8]  = (bf16_t)a2[i];
                af[i + 12] = (bf16_t)a3[i];
            }
            int c0 = wn * 32 + (lane & 15);
            v16bf b0 = mk16(*(const v8bf*)&Bs[c0][kh],      *(const v8bf*)&Bs[c0][16 + kh]);
            v16bf b1 = mk16(*(const v8bf*)&Bs[c0 + 16][kh], *(const v8bf*)&Bs[c0 + 16][16 + kh]);
            acc0 = wmma_bf16(af, b0, acc0);
            acc1 = wmma_bf16(af, b1, acc1);
        }
        __syncthreads();
    }
    int col = n0 + wn * 32 + (lane & 15);
#pragma unroll
    for (int r = 0; r < 8; ++r) {
        int m = m0 + wm * 16 + r + ((lane >> 4) << 3);
        float bias = bkv[m];
        size_t o = ((size_t)b * C2 + m) * N_PIX;
        kv[o + col]      = (bf16_t)(acc0[r] + bias);
        kv[o + col + 16] = (bf16_t)(acc1[r] + bias);
    }
}

// ---------------------------------------------------------------- 5. logits = q @ K
__global__ __launch_bounds__(32) void logits_kernel(
    const bf16_t* __restrict__ qb, const bf16_t* __restrict__ kvb,
    float* __restrict__ logits) {
    int nt = blockIdx.x * 16, mt = blockIdx.y * 16;
    int bh = blockIdx.z, b = bh >> 2, h = bh & 3;
    int lane = threadIdx.x;
    const bf16_t* qbase = qb + (size_t)b * CLS_DIM * C_DIM + h * DH;
    const bf16_t* kbase = kvb + ((size_t)b * C2 + h * DH) * N_PIX;
    v8f acc = {};
    for (int k0 = 0; k0 < DH; k0 += 32) {
        v16bf a = fragK_bf16_clamp(qbase, C_DIM, mt, CLS_DIM - 1, k0, lane);
        v16bf bb = fragS_bf16(kbase, N_PIX, nt, k0, lane);   // K[d, p]: d is K-dim (strided)
        acc = wmma_bf16(a, bb, acc);
    }
    int col = nt + (lane & 15);
#pragma unroll
    for (int r = 0; r < 8; ++r) {
        int row = mt + r + ((lane >> 4) << 3);
        if (row < CLS_DIM)
            logits[((size_t)bh * CLS_DIM + row) * N_PIX + col] = acc[r];
    }
}

// ---------------------------------------------------------------- 6. softmax (rows of 1024)
__global__ __launch_bounds__(256) void softmax_kernel(
    const float* __restrict__ logits, bf16_t* __restrict__ attn) {
    int row = blockIdx.x;
    const float* p = logits + (size_t)row * N_PIX;
    int tid = threadIdx.x;
    float v[4];
#pragma unroll
    for (int i = 0; i < 4; ++i) v[i] = p[tid + 256 * i];
    __shared__ float sh[256];
    float mx = fmaxf(fmaxf(v[0], v[1]), fmaxf(v[2], v[3]));
    sh[tid] = mx; __syncthreads();
    for (int d = 128; d > 0; d >>= 1) { if (tid < d) sh[tid] = fmaxf(sh[tid], sh[tid + d]); __syncthreads(); }
    mx = sh[0]; __syncthreads();
    float s = 0.f;
#pragma unroll
    for (int i = 0; i < 4; ++i) { v[i] = __expf(v[i] - mx); s += v[i]; }
    sh[tid] = s; __syncthreads();
    for (int d = 128; d > 0; d >>= 1) { if (tid < d) sh[tid] += sh[tid + d]; __syncthreads(); }
    float inv = 1.0f / sh[0];
    bf16_t* op = attn + (size_t)row * N_PIX;
#pragma unroll
    for (int i = 0; i < 4; ++i) op[tid + 256 * i] = (bf16_t)(v[i] * inv);
}

// ---------------------------------------------------------------- 7. out = attn @ V
__global__ __launch_bounds__(32) void attnout_kernel(
    const bf16_t* __restrict__ attn, const bf16_t* __restrict__ kvb,
    bf16_t* __restrict__ aout) {
    int dt = blockIdx.x * 16, mt = blockIdx.y * 16;
    int bh = blockIdx.z, b = bh >> 2, h = bh & 3;
    int lane = threadIdx.x;
    const bf16_t* abase = attn + (size_t)bh * CLS_DIM * N_PIX;
    const bf16_t* vbase = kvb + ((size_t)b * C2 + C_DIM + h * DH) * N_PIX;
    v8f acc = {};
    for (int k0 = 0; k0 < N_PIX; k0 += 32) {
        v16bf a = fragK_bf16_clamp(abase, N_PIX, mt, CLS_DIM - 1, k0, lane);
        v16bf bb = fragK_bf16(vbase, N_PIX, dt, k0, lane);   // V[d, p]: p is K-dim (contiguous)
        acc = wmma_bf16(a, bb, acc);
    }
    int col = dt + (lane & 15);
#pragma unroll
    for (int r = 0; r < 8; ++r) {
        int row = mt + r + ((lane >> 4) << 3);
        if (row < CLS_DIM)
            aout[((size_t)(b * CLS_DIM + row)) * C_DIM + h * DH + col] = (bf16_t)acc[r];
    }
}

// ---------------------------------------------------------------- 8. out_cls = x_cls + out @ Wp^T + bp
__global__ __launch_bounds__(32) void outcls_gemm_kernel(
    const bf16_t* __restrict__ aob, const float* __restrict__ Wp,
    const float* __restrict__ bp, const float* __restrict__ x_cls,
    float* __restrict__ out_f32, bf16_t* __restrict__ out_bf) {
    int jt = blockIdx.x * 16, mt = blockIdx.y * 16;
    int lane = threadIdx.x;
    v8f acc = {};
    for (int k0 = 0; k0 < C_DIM; k0 += 32) {
        v16bf a = fragK_bf16(aob, C_DIM, mt, k0, lane);
        v16bf b = fragK_f32(Wp, C_DIM, jt, k0, lane);
        acc = wmma_bf16(a, b, acc);
    }
    int col = jt + (lane & 15);
    float bias = bp[col];
#pragma unroll
    for (int r = 0; r < 8; ++r) {
        int row = mt + r + ((lane >> 4) << 3);
        float v = acc[r] + bias + x_cls[(size_t)row * C_DIM + col];
        out_f32[(size_t)row * C_DIM + col] = v;
        out_bf[(size_t)row * C_DIM + col] = (bf16_t)v;
    }
}

// ---------------------------------------------------------------- 9. out_patch = out_cls @ patch_flat
__global__ __launch_bounds__(32) void outpatch_kernel(
    const bf16_t* __restrict__ oclsb, const float* __restrict__ xpatch,
    float* __restrict__ opatch) {
    int nt = blockIdx.x * 16, mt = blockIdx.y * 16;
    int b = blockIdx.z;
    int lane = threadIdx.x;
    const bf16_t* abase = oclsb + (size_t)b * CLS_DIM * C_DIM;
    const float* bbase = xpatch + (size_t)b * C_DIM * N_PIX;
    v8f acc = {};
    for (int k0 = 0; k0 < C_DIM; k0 += 32) {
        v16bf a = fragK_bf16_clamp(abase, C_DIM, mt, CLS_DIM - 1, k0, lane);
        v16bf bb = fragS_f32(bbase, N_PIX, nt, k0, lane);    // patch[c, p]: c is K-dim (strided)
        acc = wmma_bf16(a, bb, acc);
    }
    int col = nt + (lane & 15);
#pragma unroll
    for (int r = 0; r < 8; ++r) {
        int row = mt + r + ((lane >> 4) << 3);
        if (row < CLS_DIM)
            opatch[((size_t)(b * CLS_DIM + row)) * N_PIX + col] = acc[r];
    }
}

// ---------------------------------------------------------------- 10. conv3x3 + bias + BN + GELU
// Per-block weight slab staged through the ASYNC path (global->LDS, no VGPRs).
__global__ __launch_bounds__(256) void conv_bn_gelu_kernel(
    const float* __restrict__ op, const float* __restrict__ Wc,
    const float* __restrict__ bconv, const float* __restrict__ g,
    const float* __restrict__ be, const float* __restrict__ bm,
    const float* __restrict__ bv, float* __restrict__ y) {
    int bo = blockIdx.x, b = bo / C_DIM, o = bo % C_DIM;
    __shared__ __align__(16) float wsm[CLS_DIM * 9];
    if (threadIdx.x < CLS_DIM * 9) {
        unsigned lds_off = (unsigned)(unsigned long long)(const void*)&wsm[threadIdx.x];
        unsigned long long ga = (unsigned long long)(const void*)
            (Wc + (size_t)o * CLS_DIM * 9 + threadIdx.x);
        asm volatile("global_load_async_to_lds_b32 %0, %1, off"
                     :: "v"(lds_off), "v"(ga) : "memory");
    }
    asm volatile("s_wait_asynccnt 0x0" ::: "memory");
    __syncthreads();
    float gg = g[o], bb = be[o], mm = bm[o], bc = bconv[o];
    float inv = rsqrtf(bv[o] + EPSF);
    const float* ip = op + (size_t)b * CLS_DIM * N_PIX;
    for (int p = threadIdx.x; p < N_PIX; p += 256) {
        int h = p >> 5, w = p & 31;
        float acc = bc;
        for (int i = 0; i < CLS_DIM; ++i) {
            const float* ch = ip + i * N_PIX;
            const float* wk = wsm + i * 9;
#pragma unroll
            for (int kh = 0; kh < 3; ++kh) {
                int hh = h + kh - 1; if ((unsigned)hh >= 32u) continue;
#pragma unroll
                for (int kw = 0; kw < 3; ++kw) {
                    int ww = w + kw - 1; if ((unsigned)ww >= 32u) continue;
                    acc += ch[hh * 32 + ww] * wk[kh * 3 + kw];
                }
            }
        }
        float z = (acc - mm) * inv * gg + bb;
        float out = 0.5f * z * (1.0f + erff(z * 0.70710678118654752f));
        y[((size_t)b * C_DIM + o) * N_PIX + p] = out;
    }
}

// ---------------------------------------------------------------- launch
extern "C" void kernel_launch(void* const* d_in, const int* in_sizes, int n_in,
                              void* d_out, int out_size, void* d_ws, size_t ws_size,
                              hipStream_t stream) {
    const float* x_cls   = (const float*)d_in[0];
    const float* xpatch  = (const float*)d_in[1];
    const float* ln_w    = (const float*)d_in[2];
    const float* ln_b    = (const float*)d_in[3];
    const float* nx_w    = (const float*)d_in[4];
    const float* nx_b    = (const float*)d_in[5];
    const float* Wq      = (const float*)d_in[6];
    const float* bq      = (const float*)d_in[7];
    const float* Wkv     = (const float*)d_in[8];
    const float* bkv     = (const float*)d_in[9];
    const float* Wp      = (const float*)d_in[10];
    const float* bp      = (const float*)d_in[11];
    const float* Wconv   = (const float*)d_in[12];
    const float* bconv   = (const float*)d_in[13];
    const float* bn_g    = (const float*)d_in[14];
    const float* bn_b    = (const float*)d_in[15];
    const float* bn_m    = (const float*)d_in[16];
    const float* bn_v    = (const float*)d_in[17];

    char* ws = (char*)d_ws;
    size_t off = 0;
    float*  stats    = (float*)(ws + off);  off += 256;                                  // 16*2 f32
    float*  part     = (float*)(ws + off);  off += 16 * 32 * 2 * sizeof(float);          // partials
    bf16_t* lnx_bf   = (bf16_t*)(ws + off); off += (size_t)320 * C_DIM * 2;              // LN(x_cls)
    bf16_t* q_bf     = (bf16_t*)(ws + off); off += (size_t)320 * C_DIM * 2;              // q (scaled)
    bf16_t* kv_bf    = (bf16_t*)(ws + off); off += (size_t)B_DIM * C2 * N_PIX * 2;       // k|v
    float*  logits   = (float*)(ws + off);  off += (size_t)B_DIM * HEADS * CLS_DIM * N_PIX * 4;
    bf16_t* attn_bf  = (bf16_t*)(ws + off); off += (size_t)B_DIM * HEADS * CLS_DIM * N_PIX * 2;
    bf16_t* aout_bf  = (bf16_t*)(ws + off); off += (size_t)320 * C_DIM * 2;              // attn out
    bf16_t* ocls_bf  = (bf16_t*)(ws + off); off += (size_t)320 * C_DIM * 2;              // out_cls bf16
    float*  opatch   = (float*)(ws + off);  off += (size_t)B_DIM * CLS_DIM * N_PIX * 4;  // out_patch

    float* out_cls_f32 = (float*)d_out;                       // first 16*20*768 floats
    float* y_f32       = (float*)d_out + (size_t)B_DIM * CLS_DIM * C_DIM;

    // 1. per-batch mean/rstd of input_patch
    stats_partial_kernel<<<dim3(32, B_DIM), 256, 0, stream>>>(xpatch, part);
    stats_final_kernel<<<B_DIM, 32, 0, stream>>>(part, stats);
    // 2. LayerNorm(x_cls) -> bf16
    ln_cls_kernel<<<B_DIM * CLS_DIM, 256, 0, stream>>>(x_cls, ln_w, ln_b, lnx_bf);
    // 3. q projection (scale folded)
    q_gemm_kernel<<<dim3(C_DIM / 16, (B_DIM * CLS_DIM) / 16), 32, 0, stream>>>(lnx_bf, Wq, bq, q_bf);
    // 4. kv = Wkv @ LN_general(input_patch)   [dominant GEMM: TDM A-tile + LDS]
    kv_gemm_kernel<<<dim3(N_PIX / 64, C2 / 64, B_DIM), 256, 0, stream>>>(
        xpatch, Wkv, bkv, nx_w, nx_b, stats, kv_bf);
    // 5. logits
    logits_kernel<<<dim3(N_PIX / 16, 2, B_DIM * HEADS), 32, 0, stream>>>(q_bf, kv_bf, logits);
    // 6. softmax
    softmax_kernel<<<B_DIM * HEADS * CLS_DIM, 256, 0, stream>>>(logits, attn_bf);
    // 7. attn @ V
    attnout_kernel<<<dim3(DH / 16, 2, B_DIM * HEADS), 32, 0, stream>>>(attn_bf, kv_bf, aout_bf);
    // 8. out_cls = x_cls + out @ Wp^T + bp  (fp32 to d_out, bf16 copy for next GEMM)
    outcls_gemm_kernel<<<dim3(C_DIM / 16, (B_DIM * CLS_DIM) / 16), 32, 0, stream>>>(
        aout_bf, Wp, bp, x_cls, out_cls_f32, ocls_bf);
    // 9. out_patch = out_cls @ patch_flat
    outpatch_kernel<<<dim3(N_PIX / 16, 2, B_DIM), 32, 0, stream>>>(ocls_bf, xpatch, opatch);
    // 10. conv3x3 + bias + BN + GELU
    conv_bn_gelu_kernel<<<B_DIM * C_DIM, 256, 0, stream>>>(
        opatch, Wconv, bconv, bn_g, bn_b, bn_m, bn_v, y_f32);
}
